// MultiHeadAttention_1331439862355
// MI455X (gfx1250) — matrix operand order repaired and checked by reading, not compiled, not run on hardware
//
#include <hip/hip_runtime.h>

typedef __attribute__((ext_vector_type(16))) _Float16 v16h;
typedef __attribute__((ext_vector_type(8)))  _Float16 v8h;
typedef __attribute__((ext_vector_type(4)))  _Float16 v4h;
typedef __attribute__((ext_vector_type(8)))  float    v8f;
typedef __attribute__((ext_vector_type(4)))  float    f32x4;
typedef __attribute__((ext_vector_type(4)))  int      v4i;

#define D_MODEL   1024
#define NUM_HEADS 16
#define DK        64
#define BATCH     2
#define SEQ       2048
#define ROWS      (BATCH * SEQ)   // 4096

// ---------------------------------------------------------------------------
// gfx1250 async global->LDS copy (ASYNCcnt-tracked, bypasses VGPRs).
// Toolchain signature (from probe diagnostic): parameters are generic v4i*
// (pointee width encodes b128), imm offset, imm cpol.
// Falls back to a plain 16B VGPR round-trip if the builtin is unavailable.
#if __has_builtin(__builtin_amdgcn_global_load_async_to_lds_b128)
#define HAVE_ASYNC_LDS 1
__device__ __forceinline__ void async_copy16(_Float16* lds_dst,
                                             const _Float16* gsrc) {
  __builtin_amdgcn_global_load_async_to_lds_b128((v4i*)gsrc, (v4i*)lds_dst,
                                                 0, 0);
}
#else
#define HAVE_ASYNC_LDS 0
__device__ __forceinline__ void async_copy16(_Float16* lds_dst,
                                             const _Float16* gsrc) {
  *(v8h*)lds_dst = *(const v8h*)gsrc;
}
#endif

__device__ __forceinline__ void wait_async() {
#if HAVE_ASYNC_LDS
#if __has_builtin(__builtin_amdgcn_s_wait_asynccnt)
  __builtin_amdgcn_s_wait_asynccnt(0);
#else
  asm volatile("s_wait_asynccnt 0x0" ::: "memory");
#endif
#endif
}

// ---------------------------------------------------------------------------
// WMMA 16x16x32 f16 fragment loaders (wave32 layouts, CDNA5 ISA).
// A fragment (16x32): lane L row M=L&15; element i: K=(L>>4)*16+i (this code's k order, matching B).
__device__ __forceinline__ v16h load_frag_a(const _Float16* __restrict__ base,
                                            int row0, int ld, int k0, int lane) {
  const int r = lane & 15, h = lane >> 4;
  const _Float16* p = base + (size_t)(row0 + r) * ld + k0 + h * 16;
  v8h lo = *(const v8h*)(p);
  v8h hi = *(const v8h*)(p + 8);
  v16h f;
#pragma unroll
  for (int i = 0; i < 8; ++i) { f[i] = lo[i]; f[i + 8] = hi[i]; }
  return f;
}

// B fragment (32x16 KxN) from an N-major buffer Bt[n][k]: lane L column
// N=L&15; element i: K=(L>>4)*16+i  -> one contiguous 16-half load.
__device__ __forceinline__ v16h load_frag_b(const _Float16* __restrict__ baseT,
                                            int col0, int ld, int k0, int lane) {
  const int n = lane & 15, h = lane >> 4;
  const _Float16* p = baseT + (size_t)(col0 + n) * ld + k0 + h * 16;
  return *(const v16h*)(p);
}

__device__ __forceinline__ v8f wmma_f16(v16h a, v16h b, v8f c) {
  return __builtin_amdgcn_wmma_f32_16x16x32_f16(false, a, false, b,
                                                (short)0, c, false, false);
}

// ---------------------------------------------------------------------------
// Elementwise fp32 -> f16 convert (4 elements/thread, vectorized).
__global__ __launch_bounds__(256) void f32_to_f16_kernel(
    const float* __restrict__ src, _Float16* __restrict__ dst) {
  const size_t i = ((size_t)blockIdx.x * 256 + threadIdx.x) * 4;
  f32x4 x = *(const f32x4*)(src + i);
  v4h y;
#pragma unroll
  for (int j = 0; j < 4; ++j) y[j] = (_Float16)x[j];
  *(v4h*)(dst + i) = y;
}

// Convert + transpose weight: w[k][n] fp32 -> wt[n][k] f16 (32x32 LDS tile).
__global__ __launch_bounds__(256) void conv_w_transpose_kernel(
    const float* __restrict__ w, _Float16* __restrict__ wt) {
  __shared__ float tile[32][33];
  const int n0 = blockIdx.x * 32, k0 = blockIdx.y * 32;
  const int c = threadIdx.x & 31, r8 = threadIdx.x >> 5;  // 8 rows/pass
#pragma unroll
  for (int rr = 0; rr < 32; rr += 8)
    tile[rr + r8][c] = w[(size_t)(k0 + rr + r8) * D_MODEL + n0 + c];
  __syncthreads();
#pragma unroll
  for (int rr = 0; rr < 32; rr += 8)
    wt[(size_t)(n0 + rr + r8) * D_MODEL + k0 + c] = (_Float16)tile[c][rr + r8];
}

// ---------------------------------------------------------------------------
// Tiled GEMM: C[4096,1024] = A[4096,1024](f16) x Wt^T + bias.
// Wt is N-major f16 [N][K] so both tiles stage as straight row-major copies ->
// async global->LDS, double-buffered so copies fly under the WMMAs.
// Each 256-thread (8-wave) workgroup owns a 128x128 tile; each wave a 32x64
// sub-tile (2x4 WMMA tiles); K staged 32 at a time.
// OUT_MODE 0: fp32 row-major out (+bias,*scale).
// OUT_MODE 1: f16 head-scatter out [B,H,T,DK] (+bias,*scale).
template <int OUT_MODE>
__global__ __launch_bounds__(256) void gemm128_wmma(
    const _Float16* __restrict__ A, const _Float16* __restrict__ Wt,
    const float* __restrict__ bias, void* __restrict__ Cp, float scale) {
  __shared__ _Float16 As[2][128 * 32];
  __shared__ _Float16 Bs[2][128 * 32];

  const int tid = threadIdx.x;
  const int lane = tid & 31, wid = tid >> 5;
  const int m0 = blockIdx.y * 128, n0 = blockIdx.x * 128;
  const int wr = wid & 3;   // 4 x 32-row wave blocks
  const int wc = wid >> 2;  // 2 x 64-col wave blocks
  const int K = D_MODEL, N = D_MODEL;

  // Stage one 128x32 K-slice of A and Wt into buffer `buf`.
  // 512 16B chunks per matrix, 2 per thread each.
  auto stage = [&](int buf, int k0) {
#pragma unroll
    for (int it = 0; it < 2; ++it) {
      const int c = tid + it * 256;
      const int r = c >> 2, kk = (c & 3) * 8;
      async_copy16(&As[buf][r * 32 + kk], A + (size_t)(m0 + r) * K + k0 + kk);
      async_copy16(&Bs[buf][r * 32 + kk], Wt + (size_t)(n0 + r) * K + k0 + kk);
    }
  };

  v8f acc[2][4];
#pragma unroll
  for (int i = 0; i < 2; ++i)
#pragma unroll
    for (int j = 0; j < 4; ++j) { v8f z = {}; acc[i][j] = z; }

  stage(0, 0);
  wait_async();
  __syncthreads();

  int buf = 0;
#pragma unroll 1
  for (int k0 = 0; k0 < K; k0 += 32) {
    if (k0 + 32 < K) stage(buf ^ 1, k0 + 32);  // async fill of next buffer

    v16h a[2], b[4];
#pragma unroll
    for (int i = 0; i < 2; ++i)
      a[i] = load_frag_a(As[buf], wr * 32 + i * 16, 32, 0, lane);
#pragma unroll
    for (int j = 0; j < 4; ++j)
      b[j] = load_frag_b(Bs[buf], wc * 64 + j * 16, 32, 0, lane);
#pragma unroll
    for (int i = 0; i < 2; ++i)
#pragma unroll
      for (int j = 0; j < 4; ++j) acc[i][j] = wmma_f16(a[i], b[j], acc[i][j]);

    wait_async();     // this wave's async fills of buf^1 have landed
    __syncthreads();  // everyone done reading buf / done filling buf^1
    buf ^= 1;
  }

  // Epilogue. C layout: lane L, vgpr v -> row v + 8*(L>>4), col L&15.
  const int hh = lane >> 4, ln = lane & 15;
#pragma unroll
  for (int i = 0; i < 2; ++i)
#pragma unroll
    for (int j = 0; j < 4; ++j)
#pragma unroll
      for (int v = 0; v < 8; ++v) {
        int row = m0 + wr * 32 + i * 16 + v + 8 * hh;
        int col = n0 + wc * 64 + j * 16 + ln;
        float val = (acc[i][j][v] + bias[col]) * scale;
        if (OUT_MODE == 0) {
          ((float*)Cp)[(size_t)row * N + col] = val;
        } else {
          int b_ = row >> 11, t = row & (SEQ - 1);
          int h_ = col >> 6, d = col & (DK - 1);
          ((_Float16*)Cp)[((((size_t)b_ * NUM_HEADS + h_) * SEQ) + t) * DK + d] =
              (_Float16)val;
        }
      }
}

// ---------------------------------------------------------------------------
// Causal flash attention. Grid: (T/128 q-blocks, B*H). 8 waves; wave w owns 16
// q rows, Q fragments pinned in registers. 32-key blocks looped only to the
// causal diagonal; K tile staged via async global->LDS; online softmax with
// 16-lane row reductions matching the C-matrix layout.
__global__ __launch_bounds__(256) void attn_wmma(
    const _Float16* __restrict__ qh, const _Float16* __restrict__ kh,
    const _Float16* __restrict__ vh, _Float16* __restrict__ ctx) {
  __shared__ _Float16 Ks[32 * 64];      // K block, key-major (== Bt for S)
  __shared__ _Float16 Vts[64 * 32];     // V block transposed (== Bt for ctx)
  __shared__ _Float16 Ps[8 * 16 * 32];  // per-wave P tile [16][32]

  const int tid = threadIdx.x, lane = tid & 31, wid = tid >> 5;
  const int hh = lane >> 4, ln = lane & 15;
  const int qb = blockIdx.x;  // q block of 128 rows
  const int bh = blockIdx.y;  // b*NUM_HEADS + h
  const size_t base = (size_t)bh * SEQ * DK;
  const int q0 = qb * 128 + wid * 16;

  v16h aq[2];
#pragma unroll
  for (int c = 0; c < 2; ++c) aq[c] = load_frag_a(qh + base, q0, DK, c * 32, lane);

  v8f o[4];
#pragma unroll
  for (int nf = 0; nf < 4; ++nf) { v8f z = {}; o[nf] = z; }
  float m_i[8], l_i[8];
#pragma unroll
  for (int v = 0; v < 8; ++v) { m_i[v] = -1e30f; l_i[v] = 0.0f; }

  _Float16* Pw = Ps + wid * (16 * 32);
  const int kb_max = qb * 4 + 3;

#pragma unroll 1
  for (int kb = 0; kb <= kb_max; ++kb) {
    const int kbase = kb * 32;
    const int key = tid >> 3, dc = (tid & 7) * 8;
    // K block (32x64): one async 16B chunk per thread.
    async_copy16(&Ks[key * 64 + dc],
                 kh + base + (size_t)(kbase + key) * DK + dc);
    // V block transposed: vector load + scalar transpose stores.
    {
      v8h vv = *(const v8h*)&vh[base + (size_t)(kbase + key) * DK + dc];
#pragma unroll
      for (int j = 0; j < 8; ++j) Vts[(dc + j) * 32 + key] = vv[j];
    }
    wait_async();
    __syncthreads();

    // S = Q x K^T : 16x32 per wave.
    v8f s[2];
#pragma unroll
    for (int j = 0; j < 2; ++j) { v8f z = {}; s[j] = z; }
#pragma unroll
    for (int c = 0; c < 2; ++c)
#pragma unroll
      for (int j = 0; j < 2; ++j)
        s[j] = wmma_f16(aq[c], load_frag_b(Ks, j * 16, DK, c * 32, lane), s[j]);

    // Causal mask + online softmax per owned row (v + 8*hh).
#pragma unroll
    for (int v = 0; v < 8; ++v) {
      const int row = q0 + v + 8 * hh;
      float mx = -1e30f;
      float sv[2];
#pragma unroll
      for (int j = 0; j < 2; ++j) {
        int col = kbase + j * 16 + ln;
        float x = s[j][v];
        if (col > row) x = -1e30f;  // causal
        sv[j] = x;
        mx = fmaxf(mx, x);
      }
#pragma unroll
      for (int off = 1; off < 16; off <<= 1)
        mx = fmaxf(mx, __shfl_xor(mx, off, 16));
      const float mnew = fmaxf(m_i[v], mx);
      const float alpha = __expf(m_i[v] - mnew);
      float p0 = __expf(sv[0] - mnew);
      float p1 = __expf(sv[1] - mnew);
      float rs = p0 + p1;
#pragma unroll
      for (int off = 1; off < 16; off <<= 1) rs += __shfl_xor(rs, off, 16);
      l_i[v] = l_i[v] * alpha + rs;
      m_i[v] = mnew;
#pragma unroll
      for (int nf = 0; nf < 4; ++nf) o[nf][v] *= alpha;
      Pw[(v + 8 * hh) * 32 + ln] = (_Float16)p0;
      Pw[(v + 8 * hh) * 32 + 16 + ln] = (_Float16)p1;
    }

    // ctx += P x V : 16x64 per wave.
    v16h ap = load_frag_a(Pw, 0, 32, 0, lane);
#pragma unroll
    for (int nf = 0; nf < 4; ++nf)
      o[nf] = wmma_f16(ap, load_frag_b(Vts, nf * 16, 32, 0, lane), o[nf]);
    __syncthreads();
  }

  // Normalize; write ctx as [B, T, H*DK] row-major f16 (GEMM-ready).
  const int b_ = bh >> 4, h_ = bh & (NUM_HEADS - 1);
#pragma unroll
  for (int nf = 0; nf < 4; ++nf)
#pragma unroll
    for (int v = 0; v < 8; ++v) {
      int row = q0 + v + 8 * hh;
      int d = nf * 16 + ln;
      float val = o[nf][v] / l_i[v];
      ctx[((size_t)(b_ * SEQ + row)) * D_MODEL + h_ * DK + d] = (_Float16)val;
    }
}

// ---------------------------------------------------------------------------
extern "C" void kernel_launch(void* const* d_in, const int* in_sizes, int n_in,
                              void* d_out, int out_size, void* d_ws, size_t ws_size,
                              hipStream_t stream) {
  (void)in_sizes; (void)n_in; (void)out_size; (void)ws_size;
  const float* q  = (const float*)d_in[0];
  const float* k  = (const float*)d_in[1];
  const float* v  = (const float*)d_in[2];
  // d_in[3] = causal mask: handled analytically in attn_wmma.
  const float* wq = (const float*)d_in[4];
  const float* bq = (const float*)d_in[5];
  const float* wk = (const float*)d_in[6];
  const float* bk = (const float*)d_in[7];
  const float* wv = (const float*)d_in[8];
  const float* bv = (const float*)d_in[9];
  const float* wo = (const float*)d_in[10];
  const float* bo = (const float*)d_in[11];
  float* out = (float*)d_out;

  // Workspace (f16): Xh 8MiB (reused per projection) | WT 2MiB (reused) |
  // qh,kh,vh 8MiB each | ctx 8MiB  => 42 MiB total.
  const size_t XY = (size_t)ROWS * D_MODEL;      // 4096*1024
  const size_t WW = (size_t)D_MODEL * D_MODEL;   // 1024*1024
  _Float16* Xh  = (_Float16*)d_ws;
  _Float16* WT  = Xh + XY;
  _Float16* qh  = WT + WW;
  _Float16* khd = qh + XY;
  _Float16* vhd = khd + XY;
  _Float16* ctx = vhd + XY;

  dim3 blk(256);
  dim3 cvt_grid((unsigned)(XY / (256 * 4)));          // 4096 blocks
  dim3 wt_grid(D_MODEL / 32, D_MODEL / 32);           // (32, 32)
  dim3 gemm_grid(D_MODEL / 128, ROWS / 128);          // (8, 32)
  dim3 attn_grid(SEQ / 128, BATCH * NUM_HEADS);       // (16, 32)
  const float inv_sqrt_dk = 0.125f;  // 1/sqrt(64), folded into Q

  // Q projection
  f32_to_f16_kernel<<<cvt_grid, blk, 0, stream>>>(q, Xh);
  conv_w_transpose_kernel<<<wt_grid, blk, 0, stream>>>(wq, WT);
  gemm128_wmma<1><<<gemm_grid, blk, 0, stream>>>(Xh, WT, bq, qh, inv_sqrt_dk);
  // K projection
  f32_to_f16_kernel<<<cvt_grid, blk, 0, stream>>>(k, Xh);
  conv_w_transpose_kernel<<<wt_grid, blk, 0, stream>>>(wk, WT);
  gemm128_wmma<1><<<gemm_grid, blk, 0, stream>>>(Xh, WT, bk, khd, 1.0f);
  // V projection
  f32_to_f16_kernel<<<cvt_grid, blk, 0, stream>>>(v, Xh);
  conv_w_transpose_kernel<<<wt_grid, blk, 0, stream>>>(wv, WT);
  gemm128_wmma<1><<<gemm_grid, blk, 0, stream>>>(Xh, WT, bv, vhd, 1.0f);
  // Attention
  attn_wmma<<<attn_grid, blk, 0, stream>>>(qh, khd, vhd, ctx);
  // Output projection
  conv_w_transpose_kernel<<<wt_grid, blk, 0, stream>>>(wo, WT);
  gemm128_wmma<0><<<gemm_grid, blk, 0, stream>>>(ctx, WT, bo, out, 1.0f);
}
